// LearnedInterventionAttribution_52888227283425
// MI455X (gfx1250) — matrix operand order repaired and checked by reading, not compiled, not run on hardware
//
#include <hip/hip_runtime.h>
#include <hip/hip_bf16.h>

// Problem constants (from reference)
#define BATCH   32
#define SEQ     4096
#define HID     768
#define DIM     192          // HID/4
#define NQ      16           // 15 signals padded to 16
#define TCHUNK  128          // tokens per workgroup
#define NCHUNK  (SEQ / TCHUNK)          // 32 chunks per batch
#define HC      32           // hidden elements per pipeline stage (one WMMA k-step)

typedef __attribute__((ext_vector_type(16))) __bf16 v16bf;
typedef __attribute__((ext_vector_type(8)))  __bf16 v8bf;
typedef __attribute__((ext_vector_type(8)))  float  v8f;
typedef __attribute__((ext_vector_type(4)))  unsigned int u32x4;
typedef __attribute__((ext_vector_type(8)))  int  i32x8;
typedef __attribute__((ext_vector_type(4)))  int  i32x4;

// ---- LDS layout for k_attn_partial (dynamic, bytes) ---------------------
//   [0      ) Kt : 128x192 bf16 = 49152   (K, row-major by token)
//   [49152  ) Vt : 192x128 bf16 = 49152   (V, row-major by d)
//   [98304  ) Qs : 16x192  bf16 =  6144
//   [104448 ) Xf : 2 x (128x32 f32) = 32768    (TDM double buffer, fp32)
//       phase-B reuse: Sc 16x128 f32 (8192) @104448 | Pt 16x128 bf16 @112640
//                      stats @116736
//   [137216 ) Wd : 2 x (384x32 bf16) = 49152   (TDM double buffer)
//   total 186368
#define LDS_KT   0
#define LDS_VT   49152
#define LDS_QS   98304
#define LDS_XF   104448
#define LDS_SC   104448
#define LDS_PT   112640
#define LDS_ST   116736
#define LDS_WD   137216
#define LDS_TOTAL 186368

// ---- WMMA operand loaders ------------------------------------------------
// A-matrix 16x32 bf16: lanes 0-15 hold K=0..15, lanes 16-31 hold K=16..31
// (two contiguous 8-element runs).
__device__ __forceinline__ v16bf ldA16(const __bf16* p) {
  v8bf lo = *(const v8bf*)(p);
  v8bf hi = *(const v8bf*)(p + 8);
  v16bf r;
#pragma unroll
  for (int i = 0; i < 8; ++i) { r[i] = lo[i]; r[i + 8] = hi[i]; }
  return r;
}
// Same A layout, but source is fp32 in LDS (TDM-staged X): convert on load.
__device__ __forceinline__ v16bf ldA32cvt(const float* p) {
  float4 a0 = *(const float4*)(p);
  float4 a1 = *(const float4*)(p + 4);
  float4 b0 = *(const float4*)(p + 8);
  float4 b1 = *(const float4*)(p + 12);
  v16bf r;
  r[0]  = (__bf16)a0.x; r[1]  = (__bf16)a0.y; r[2]  = (__bf16)a0.z; r[3]  = (__bf16)a0.w;
  r[4]  = (__bf16)a1.x; r[5]  = (__bf16)a1.y; r[6]  = (__bf16)a1.z; r[7]  = (__bf16)a1.w;
  r[8]  = (__bf16)b0.x; r[9]  = (__bf16)b0.y; r[10] = (__bf16)b0.z; r[11] = (__bf16)b0.w;
  r[12] = (__bf16)b1.x; r[13] = (__bf16)b1.y; r[14] = (__bf16)b1.z; r[15] = (__bf16)b1.w;
  return r;
}
// B-matrix 32x16 bf16: lanes 0-15 hold K=0..15, lanes 16-31 hold K=16..31.
__device__ __forceinline__ v16bf ldB16(const __bf16* p) {
  return *(const v16bf*)p;
}
__device__ __forceinline__ v8f wmma_bf16(v16bf a, v16bf b, v8f c) {
  return __builtin_amdgcn_wmma_f32_16x16x32_bf16(false, a, false, b,
                                                 (short)0, c, false, false);
}

// ---- Tensor Data Mover: 2D tile global->LDS (ISA ch.8 D# layout) ---------
#if __has_builtin(__builtin_amdgcn_tensor_load_to_lds) && __has_builtin(__builtin_amdgcn_s_wait_tensorcnt)
#define USE_TDM 1
#else
#define USE_TDM 0
#endif

#if USE_TDM
// ds_code: 0=1B 1=2B 2=4B 3=8B elements. All dims in elements.
__device__ __forceinline__ void tdm_load_2d(const void* gaddr, unsigned lds_off,
                                            unsigned ds_code,
                                            unsigned tile_d0, unsigned tile_d1,
                                            unsigned tensor_d0, unsigned tensor_d1,
                                            unsigned stride0) {
  unsigned long long ga = (unsigned long long)gaddr;
  u32x4 g0;
  g0[0] = 1u;                                       // count=1, user descriptor
  g0[1] = lds_off;                                  // lds_addr (bytes)
  g0[2] = (unsigned)ga;                             // global_addr[31:0]
  g0[3] = (unsigned)((ga >> 32) & 0x01FFFFFFu) | (2u << 30);  // addr[56:32] | type=2
  i32x8 g1;
  g1[0] = (int)(ds_code << 16);                     // wg_mask=0 | data_size
  g1[1] = (int)((tensor_d0 & 0xFFFFu) << 16);       // [63:48]=tensor_dim0.lo16
  g1[2] = (int)((tensor_d0 >> 16) | ((tensor_d1 & 0xFFFFu) << 16));
  g1[3] = (int)((tensor_d1 >> 16) | (tile_d0 << 16));
  g1[4] = (int)(tile_d1);                           // tile_dim1 | tile_dim2=0
  g1[5] = (int)(stride0);                           // tensor_dim0_stride[31:0]
  g1[6] = 0;                                        // stride0.hi | stride1.lo (2D: 0)
  g1[7] = 0;
  i32x4 z4 = {0, 0, 0, 0};
#if defined(__clang_major__) && (__clang_major__ >= 23)
  i32x8 z8 = {0, 0, 0, 0, 0, 0, 0, 0};
  __builtin_amdgcn_tensor_load_to_lds(g0, g1, z4, z4, z8, 0);
#else
  __builtin_amdgcn_tensor_load_to_lds(g0, g1, z4, z4, 0);
#endif
}
#endif

// =========================================================================
// Kernel 0: convert weights to bf16, transposed for contiguous staging.
//   Wt[proj][col(192)][h(768)]  bf16   (proj 0 = Wk, 1 = Wv; src is [h][col])
//   Qb[q(16)][d(192)]           bf16   (row 15 zero padding)
// =========================================================================
__global__ void k_prep(const float* __restrict__ Wk, const float* __restrict__ Wv,
                       const float* __restrict__ sq,
                       __bf16* __restrict__ Wt, __bf16* __restrict__ Qb) {
  int idx = blockIdx.x * 256 + threadIdx.x;
  const int NW = 2 * DIM * HID;            // 294912
  if (idx < NW) {
    int proj = idx / (DIM * HID);
    int rem  = idx % (DIM * HID);
    int col  = rem / HID;
    int h    = rem % HID;
    const float* W = proj ? Wv : Wk;
    Wt[idx] = (__bf16)W[h * DIM + col];
  } else if (idx < NW + NQ * DIM) {
    int i2 = idx - NW;
    int q = i2 / DIM, d = i2 % DIM;
    Qb[i2] = (__bf16)((q < 15) ? sq[q * DIM + d] : 0.0f);
  }
}

// =========================================================================
// Kernel 1: fused K/V projection + cross-attention softmax partials.
// One WG = (batch b, token chunk c) over 128 tokens, 8 waves of 32.
// Phase A wave grid: 2 rowgroups x 4 colgroups; each wave owns 4 row-tiles
// (64 tokens) x 6 (proj,col) pairs = 24 accumulator tiles.
// TDM double-buffered pipeline over 24 hidden chunks of 32.
// =========================================================================
__global__ __launch_bounds__(256, 1)
void k_attn_partial(const float* __restrict__ X,
                    const float* __restrict__ bk, const float* __restrict__ bv,
                    const __bf16* __restrict__ Wt, const __bf16* __restrict__ Qb,
                    float* __restrict__ partO, float* __restrict__ partM,
                    float* __restrict__ partZ, float* __restrict__ partT) {
  extern __shared__ char smem[];
  __bf16* Kt  = (__bf16*)(smem + LDS_KT);
  __bf16* Vt  = (__bf16*)(smem + LDS_VT);
  __bf16* Qs  = (__bf16*)(smem + LDS_QS);
  float*  Xf  = (float*) (smem + LDS_XF);   // [2][128][32] f32
  __bf16* Wd  = (__bf16*)(smem + LDS_WD);   // [2][384][32] bf16
  float*  Sc  = (float*) (smem + LDS_SC);
  __bf16* Pt  = (__bf16*)(smem + LDS_PT);
  float*  stM = (float*) (smem + LDS_ST);
  const unsigned ldsbase = (unsigned)(unsigned long long)(void*)smem;

  const int tid   = threadIdx.x;
  const int lane  = tid & 31;
  const int wave  = tid >> 5;
  const int mrow  = lane & 15;
  const int half  = (lane >> 4) & 1;
  const int kb8   = half * 8;               // unused; A operands use kb16 as well
  const int kb16  = half * 16;              // B-operand k run base
  const int rg    = wave >> 2;              // rowgroup 0..1 (4 row-tiles each)
  const int cg    = wave & 3;               // colgroup 0..3 (6 pairs each)
  const int b     = blockIdx.y;
  const int cidx  = blockIdx.x;
  const int chunk = b * NCHUNK + cidx;
  const int s0    = cidx * TCHUNK;

  const float* Xbase = X + ((size_t)(b * SEQ + s0)) * HID;

  // stage Q once (6144 B = 384 x 16B)
  {
    const uint4* src = (const uint4*)Qb;
    uint4* dst = (uint4*)Qs;
    for (int i = tid; i < 384; i += 256) dst[i] = src[i];
  }

  // ---------------- Phase A: K = X@Wk, V = X@Wv (bf16 WMMA) --------------
  v8f acc[4][6];
#pragma unroll
  for (int rt = 0; rt < 4; ++rt)
#pragma unroll
    for (int i = 0; i < 6; ++i)
#pragma unroll
      for (int e = 0; e < 8; ++e) acc[rt][i][e] = 0.0f;

  const int pair0 = cg * 6;

#if USE_TDM
  // Prologue: wave 0 DMAs chunk 0 into buffer 0.
  if (wave == 0) {
    tdm_load_2d(Xbase,  ldsbase + LDS_XF, 2u, HC, TCHUNK, HID, SEQ, HID);
    tdm_load_2d(Wt,     ldsbase + LDS_WD, 1u, HC, 2 * DIM, HID, 2 * DIM, HID);
    __builtin_amdgcn_s_wait_tensorcnt(0);
  }
  __syncthreads();
#endif

  for (int it = 0; it < HID / HC; ++it) {
    const int h0  = it * HC;
    const int cur = it & 1;
#if USE_TDM
    // Issue DMA for the next chunk into the other buffer.
    if (wave == 0 && h0 + HC < HID) {
      const int nxt = cur ^ 1;
      tdm_load_2d(Xbase + (h0 + HC), ldsbase + LDS_XF + nxt * 16384,
                  2u, HC, TCHUNK, HID, SEQ, HID);
      tdm_load_2d(Wt + (h0 + HC),    ldsbase + LDS_WD + nxt * 24576,
                  1u, HC, 2 * DIM, HID, 2 * DIM, HID);
    }
#else
    // Fallback: cooperative staging into the current buffer.
    {
      for (int i = tid; i < 1024; i += 256) {     // X: 128x32 f32, float4
        int r = i >> 3, c4 = i & 7;
        *(float4*)(Xf + cur * 4096 + r * HC + c4 * 4) =
            *(const float4*)(Xbase + (size_t)r * HID + h0 + c4 * 4);
      }
      for (int i = tid; i < 1536; i += 256) {     // W: 384x32 bf16, 16B
        int pc = i >> 2, seg = i & 3;
        *(uint4*)(Wd + cur * 12288 + pc * HC + seg * 8) =
            *(const uint4*)(Wt + (size_t)pc * HID + h0 + seg * 8);
      }
      __syncthreads();
    }
#endif

    // Compute on current buffer: preload ALL operand tiles first so the
    // ds_loads overlap, then issue the 24 WMMAs back-to-back (no per-tile
    // load->wait->cvt->wmma serialization, no WMMA/VALU hazard nops).
    const float*  Xc = Xf + cur * 4096;
    const __bf16* Wc = Wd + cur * 12288;
    v16bf bmat[6];
#pragma unroll
    for (int i = 0; i < 6; ++i) {
      int pair = pair0 + i;
      int proj = pair / 12, ct = pair % 12;
      int ncol = ct * 16 + mrow;
      bmat[i] = ldB16(Wc + (proj * DIM + ncol) * HC + kb16);
    }
    v16bf amat[4];
#pragma unroll
    for (int rt = 0; rt < 4; ++rt)
      amat[rt] = ldA32cvt(Xc + ((rg * 4 + rt) * 16 + mrow) * HC + kb16);
#pragma unroll
    for (int rt = 0; rt < 4; ++rt)
#pragma unroll
      for (int i = 0; i < 6; ++i)
        acc[rt][i] = wmma_bf16(amat[rt], bmat[i], acc[rt][i]);

#if USE_TDM
    if (wave == 0 && h0 + HC < HID) __builtin_amdgcn_s_wait_tensorcnt(0);
#endif
    __syncthreads();   // next buffer ready / current buffer free for reuse
  }

  // bias + write K/V tiles (bf16) to LDS. C/D layout: n = lane%16, m = r + 8*half.
#pragma unroll
  for (int i = 0; i < 6; ++i) {
    int pair = pair0 + i;
    int proj = pair / 12, ct = pair % 12;
    int ncol = ct * 16 + mrow;
    float bias = proj ? bv[ncol] : bk[ncol];
#pragma unroll
    for (int rt = 0; rt < 4; ++rt) {
#pragma unroll
      for (int r = 0; r < 8; ++r) {
        int tok = (rg * 4 + rt) * 16 + r + 8 * half;
        __bf16 val = (__bf16)(acc[rt][i][r] + bias);
        if (proj == 0) Kt[tok * DIM + ncol] = val;     // [token][d]
        else           Vt[ncol * TCHUNK + tok] = val;  // [d][token]
      }
    }
  }
  __syncthreads();

  // ---------------- Phase B: scores = Q @ K^T / sqrt(d) ------------------
  {
    v8f sacc = {0.f, 0.f, 0.f, 0.f, 0.f, 0.f, 0.f, 0.f};
    int tok = wave * 16 + mrow;                 // B-operand column (token)
#pragma unroll
    for (int ks = 0; ks < 6; ++ks) {
      v16bf a  = ldA16(Qs + mrow * DIM + ks * 32 + kb16);
      v16bf bm = ldB16(Kt + tok * DIM + ks * 32 + kb16);
      sacc = wmma_bf16(a, bm, sacc);
    }
    const float scale = 0.0721687836f;          // 1/sqrt(192)
#pragma unroll
    for (int r = 0; r < 8; ++r) {
      int q = r + 8 * half;
      Sc[q * TCHUNK + wave * 16 + mrow] = sacc[r] * scale;
    }
  }
  __syncthreads();

  // per-q chunk-local softmax stats: m, Z = sum e^(l-m), T = sum e^(l-m)(l-m)
  if (tid < NQ) {
    int q = tid;
    float m = -1e30f;
    for (int t = 0; t < TCHUNK; ++t) m = fmaxf(m, Sc[q * TCHUNK + t]);
    float Z = 0.f, T = 0.f;
    for (int t = 0; t < TCHUNK; ++t) {
      float d = Sc[q * TCHUNK + t] - m;
      float e = __expf(d);
      Z += e; T += e * d;
    }
    stM[q] = m;
    partM[chunk * NQ + q] = m;
    partZ[chunk * NQ + q] = Z;
    partT[chunk * NQ + q] = T;
  }
  __syncthreads();

  // P = e^(l - m)  (unnormalized), bf16, layout [q][token]
  for (int i = tid; i < NQ * TCHUNK; i += 256) {
    int q = i >> 7, t = i & 127;
    Pt[q * TCHUNK + t] = (__bf16)__expf(Sc[q * TCHUNK + t] - stM[q]);
  }
  __syncthreads();

  // ---------------- Phase C: partial O = P @ V ---------------------------
  const int ntile = (wave < 4) ? 2 : 1;         // 12 d-tiles over 8 waves
  for (int itile = 0; itile < ntile; ++itile) {
    int dt = (itile == 0) ? wave : (8 + wave);
    int dcol = dt * 16 + mrow;
    v8f oacc = {0.f, 0.f, 0.f, 0.f, 0.f, 0.f, 0.f, 0.f};
#pragma unroll
    for (int ks = 0; ks < 4; ++ks) {
      v16bf a  = ldA16(Pt + mrow * TCHUNK + ks * 32 + kb16);
      v16bf bm = ldB16(Vt + dcol * TCHUNK + ks * 32 + kb16);
      oacc = wmma_bf16(a, bm, oacc);
    }
    size_t base = (size_t)chunk * (NQ * DIM);
#pragma unroll
    for (int r = 0; r < 8; ++r) {
      int q = r + 8 * half;
      partO[base + q * DIM + dt * 16 + mrow] = oacc[r];
    }
  }
}

// =========================================================================
// Kernel 2: per-batch LSE merge of chunk partials + entropy + GELU MLP.
// out[b][0:15] = effects, out[b][15:30] = strength.
// =========================================================================
__global__ __launch_bounds__(256)
void k_combine(const float* __restrict__ partO, const float* __restrict__ partM,
               const float* __restrict__ partZ, const float* __restrict__ partT,
               const float* __restrict__ nullE, const float* __restrict__ W1,
               const float* __restrict__ b1, const float* __restrict__ W2,
               const float* __restrict__ b2, float* __restrict__ out) {
  __shared__ float sig[15][DIM];
  __shared__ float sM[NQ], sZ[NQ];
  __shared__ float hdn[15][64];
  const int b = blockIdx.x, tid = threadIdx.x;
  const int cbase = b * NCHUNK;

  if (tid < NQ) {
    int q = tid;
    float m = -1e30f;
    for (int c = 0; c < NCHUNK; ++c) m = fmaxf(m, partM[(cbase + c) * NQ + q]);
    float Z = 0.f, T = 0.f;
    for (int c = 0; c < NCHUNK; ++c) {
      float mc = partM[(cbase + c) * NQ + q];
      float w  = __expf(mc - m);
      float Zc = partZ[(cbase + c) * NQ + q];
      Z += w * Zc;
      T += w * (partT[(cbase + c) * NQ + q] + (mc - m) * Zc);
    }
    sM[q] = m; sZ[q] = Z;
    if (q < 15) {
      float ent = __logf(Z) - T / Z;
      out[b * 30 + 15 + q] = 1.0f - ent * (1.0f / 8.3177661667f); // 1/log(4096)
    }
  }
  __syncthreads();

  for (int i = tid; i < 15 * DIM; i += 256) {
    int q = i / DIM, d = i % DIM;
    float s = 0.f;
    for (int c = 0; c < NCHUNK; ++c) {
      float w = __expf(partM[(cbase + c) * NQ + q] - sM[q]);
      s += w * partO[(size_t)(cbase + c) * (NQ * DIM) + q * DIM + d];
    }
    sig[q][d] = s / sZ[q];
  }
  __syncthreads();

  // hdn = gelu([sig, null[:, :192]] @ W1 + b1)
  for (int i = tid; i < 15 * 64; i += 256) {
    int q = i / 64, j = i % 64;
    float a = b1[j];
    for (int k = 0; k < DIM; ++k) a += sig[q][k] * W1[k * 64 + j];
    for (int k = 0; k < DIM; ++k) a += nullE[q * HID + k] * W1[(DIM + k) * 64 + j];
    hdn[q][j] = 0.5f * a * (1.0f + erff(a * 0.70710678f));
  }
  __syncthreads();

  if (tid < 15) {
    float e = b2[0];
    for (int j = 0; j < 64; ++j) e += hdn[tid][j] * W2[j];
    out[b * 30 + tid] = e;
  }
}

// =========================================================================
extern "C" void kernel_launch(void* const* d_in, const int* in_sizes, int n_in,
                              void* d_out, int out_size, void* d_ws, size_t ws_size,
                              hipStream_t stream) {
  const float* X    = (const float*)d_in[0];
  const float* Wk   = (const float*)d_in[1];
  const float* bk   = (const float*)d_in[2];
  const float* Wv   = (const float*)d_in[3];
  const float* bv   = (const float*)d_in[4];
  const float* sq   = (const float*)d_in[5];
  const float* nullE= (const float*)d_in[6];
  const float* W1   = (const float*)d_in[7];
  const float* b1   = (const float*)d_in[8];
  const float* W2   = (const float*)d_in[9];
  const float* b2   = (const float*)d_in[10];
  float* out = (float*)d_out;

  // workspace layout (~12.8 MB)
  char* ws = (char*)d_ws;
  __bf16* Wt = (__bf16*)(ws);                      // 589824 B
  __bf16* Qb = (__bf16*)(ws + 589824);             //   6144 B
  float* partO = (float*)(ws + 595968);            // 1024*16*192*4 = 12582912 B
  float* partM = (float*)(ws + 13178880);          // 65536 B
  float* partZ = (float*)(ws + 13244416);          // 65536 B
  float* partT = (float*)(ws + 13309952);          // 65536 B

  k_prep<<<dim3((2 * DIM * HID + NQ * DIM) / 256), dim3(256), 0, stream>>>(
      Wk, Wv, sq, Wt, Qb);
  k_attn_partial<<<dim3(NCHUNK, BATCH), dim3(256), LDS_TOTAL, stream>>>(
      X, bk, bv, Wt, Qb, partO, partM, partZ, partT);
  k_combine<<<dim3(BATCH), dim3(256), 0, stream>>>(
      partO, partM, partZ, partT, nullE, W1, b1, W2, b2, out);
}
